// HubertWithKmeans_20890720928672
// MI455X (gfx1250) — compile-verified, matrix operand order checked by reading, not run on hardware
//
#include <hip/hip_runtime.h>

// ---------------------------------------------------------------------------
// HubertWithKmeans argmin-distance clustering for MI455X (gfx1250, wave32).
//   clusters[m] = argmax_c ( embed[m]·center[c] - 0.5*||center[c]||^2 )
// Split-bf16 WMMA GEMM (hi*hi + hi*lo + lo*hi ~ fp32 accuracy), argmax fused.
// Each wave: 4 row-tiles x 2 col-tiles -> 24 WMMAs per 24 mem-ops per k-step.
// ---------------------------------------------------------------------------

typedef __bf16 bf16_t;
typedef __bf16 v8bf   __attribute__((ext_vector_type(8)));
typedef __bf16 v16bf  __attribute__((ext_vector_type(16)));
typedef float  v8f    __attribute__((ext_vector_type(8)));

#define M_TOT 24000        // B*T = 16*1500
#define DK    768          // feature dim (K)
#define CN    1024         // clusters (N)
#define MBLK  64           // rows per workgroup (4 WMMA row-tiles)
#define NRT   4            // row-tiles per wave
#define KSTEPS (DK / 32)   // 24 k-steps of 32 for bf16 WMMA

// Build a 16-element bf16 fragment from two contiguous 8-element (16B) loads.
// Matches the ISA 16-bit A(16x32)/B(32x16) layout: a lane holds K blocks
// {kb, kb+16} where kb = kstep*32 + (lane>=16 ? 8 : 0).
static __device__ inline v16bf make_frag(const bf16_t* p0, const bf16_t* p1) {
    v8bf a = *(const v8bf*)p0;
    v8bf b = *(const v8bf*)p1;
    return __builtin_shufflevector(a, b, 0, 1, 2, 3, 4, 5, 6, 7,
                                         8, 9, 10, 11, 12, 13, 14, 15);
}

static __device__ inline v8f wmma_bf16(v16bf a, v16bf b, v8f c) {
    return __builtin_amdgcn_wmma_f32_16x16x32_bf16(
        /*neg_a=*/false, a, /*neg_b=*/false, b,
        /*c_mod=*/(short)0, c, /*reuse_a=*/false, /*reuse_b=*/false);
}

// ---------------------------------------------------------------------------
// Prep: centers (fp32 [C][D]) -> split bf16 hi/lo arrays + 0.5*||c||^2.
// ---------------------------------------------------------------------------
__global__ __launch_bounds__(256)
void prep_centers_kernel(const float* __restrict__ centers,
                         bf16_t* __restrict__ b_hi, bf16_t* __restrict__ b_lo,
                         float* __restrict__ c_half) {
    const int c = blockIdx.x;
    const int t = threadIdx.x;
    const float* row = centers + (size_t)c * DK;
    float ss = 0.0f;
    for (int d = t; d < DK; d += 256) {
        float x  = row[d];
        bf16_t h = (bf16_t)x;          // RNE fp32 -> bf16
        float  hf = (float)h;
        bf16_t l = (bf16_t)(x - hf);   // residual
        b_hi[(size_t)c * DK + d] = h;
        b_lo[(size_t)c * DK + d] = l;
        ss += x * x;
    }
    __shared__ float red[256];
    red[t] = ss;
    __syncthreads();
    for (int s = 128; s > 0; s >>= 1) {
        if (t < s) red[t] += red[t + s];
        __syncthreads();
    }
    if (t == 0) c_half[c] = 0.5f * red[0];
}

// ---------------------------------------------------------------------------
// Main: 64 embed rows per block staged in LDS (split bf16, 192KB).
// 8 waves; each wave owns ALL 4 row-tiles and sweeps a strided eighth of the
// 1024 columns, 2 col-tiles (32 cols) per chunk => 8 accumulator tiles and
// 24 WMMAs per k-step with B fragments reused across 4 row-tiles.
// ---------------------------------------------------------------------------
__global__ __launch_bounds__(256)
void kmeans_argmax_kernel(const float*  __restrict__ embed,   // [M][DK] fp32
                          const bf16_t* __restrict__ b_hi,    // [CN][DK]
                          const bf16_t* __restrict__ b_lo,    // [CN][DK]
                          const float*  __restrict__ c_half,  // [CN]
                          int* __restrict__ out) {             // [M]
    __shared__ bf16_t a_hi[MBLK * DK];   // 96 KB
    __shared__ bf16_t a_lo[MBLK * DK];   // 96 KB
    __shared__ float  red_v[MBLK][8];
    __shared__ int    red_i[MBLK][8];

    const int tid    = threadIdx.x;
    const int wave   = tid >> 5;        // 0..7
    const int lane   = tid & 31;
    const int l16    = lane & 15;
    const int laneHi = lane >> 4;       // 0: K{0..7,16..23}, 1: K{8..15,24..31}
    const int row0   = blockIdx.x * MBLK;

    // ---- stage + split-convert the A block into LDS (float4 vectorized) ----
    {
        const float4* src = (const float4*)(embed + (size_t)row0 * DK);
        for (int i = tid; i < (MBLK * DK) / 4; i += 256) {
            float4 v = src[i];
            float xs[4] = {v.x, v.y, v.z, v.w};
            #pragma unroll
            for (int j = 0; j < 4; ++j) {
                float  x  = xs[j];
                bf16_t h  = (bf16_t)x;
                a_hi[i * 4 + j] = h;
                a_lo[i * 4 + j] = (bf16_t)(x - (float)h);
            }
        }
    }
    __syncthreads();

    // Running best per lane: slot [rt][j] covers row rt*16 + j + (laneHi?8:0).
    float bestv[NRT][8];
    int   besti[NRT][8];
    #pragma unroll
    for (int rt = 0; rt < NRT; ++rt)
        #pragma unroll
        for (int j = 0; j < 8; ++j) { bestv[rt][j] = -3.4e38f; besti[rt][j] = 0; }

    const bf16_t* ah[NRT];
    const bf16_t* al[NRT];
    #pragma unroll
    for (int rt = 0; rt < NRT; ++rt) {
        ah[rt] = a_hi + (rt * 16 + l16) * DK;
        al[rt] = a_lo + (rt * 16 + l16) * DK;
    }

    for (int ch = wave; ch < CN / 32; ch += 8) {   // 4 chunks of 32 cols
        const int nbase = ch * 32;
        const int col0  = nbase + l16;        // col-tile 0 column of this lane
        const int col1  = nbase + 16 + l16;   // col-tile 1 column
        const bf16_t* bh0 = b_hi + (size_t)col0 * DK;
        const bf16_t* bl0 = b_lo + (size_t)col0 * DK;
        const bf16_t* bh1 = b_hi + (size_t)col1 * DK;
        const bf16_t* bl1 = b_lo + (size_t)col1 * DK;

        v8f acc[NRT][2];
        #pragma unroll
        for (int rt = 0; rt < NRT; ++rt) { acc[rt][0] = (v8f){}; acc[rt][1] = (v8f){}; }

        #pragma unroll 1
        for (int ks = 0; ks < KSTEPS; ++ks) {
            const int kb0 = ks * 32 + (laneHi ? 8 : 0);
            const int kb1 = kb0 + 16;

            // A fragments: 4 row-tiles x hi/lo  (ds_load_b128 x16)
            v16bf afh[NRT], afl[NRT];
            #pragma unroll
            for (int rt = 0; rt < NRT; ++rt) {
                afh[rt] = make_frag(ah[rt] + kb0, ah[rt] + kb1);
                afl[rt] = make_frag(al[rt] + kb0, al[rt] + kb1);
            }

            // B col-tile 0 (global_load_b128 x4), reused by all 4 row-tiles.
            {
                v16bf bfh = make_frag(bh0 + kb0, bh0 + kb1);
                v16bf bfl = make_frag(bl0 + kb0, bl0 + kb1);
                #pragma unroll
                for (int rt = 0; rt < NRT; ++rt) {
                    acc[rt][0] = wmma_bf16(afh[rt], bfh, acc[rt][0]);
                    acc[rt][0] = wmma_bf16(afh[rt], bfl, acc[rt][0]);
                    acc[rt][0] = wmma_bf16(afl[rt], bfh, acc[rt][0]);
                }
            }
            // B col-tile 1
            {
                v16bf bfh = make_frag(bh1 + kb0, bh1 + kb1);
                v16bf bfl = make_frag(bl1 + kb0, bl1 + kb1);
                #pragma unroll
                for (int rt = 0; rt < NRT; ++rt) {
                    acc[rt][1] = wmma_bf16(afh[rt], bfh, acc[rt][1]);
                    acc[rt][1] = wmma_bf16(afh[rt], bfl, acc[rt][1]);
                    acc[rt][1] = wmma_bf16(afl[rt], bfh, acc[rt][1]);
                }
            }
        }

        // Fused argmax epilogue: score = cross - 0.5*||c||^2.
        const float chalf0 = c_half[col0];
        const float chalf1 = c_half[col1];
        #pragma unroll
        for (int rt = 0; rt < NRT; ++rt) {
            #pragma unroll
            for (int j = 0; j < 8; ++j) {
                float s;
                s = acc[rt][0][j] - chalf0;
                if (s > bestv[rt][j]) { bestv[rt][j] = s; besti[rt][j] = col0; }
                s = acc[rt][1][j] - chalf1;
                if (s > bestv[rt][j]) { bestv[rt][j] = s; besti[rt][j] = col1; }
            }
        }
    }

    // Butterfly argmax across the 16 lanes sharing a row (C-tile layout:
    // VGPR j -> row j (lanes 0-15) / row j+8 (lanes 16-31); N = lane&15).
    #pragma unroll
    for (int rt = 0; rt < NRT; ++rt) {
        #pragma unroll
        for (int j = 0; j < 8; ++j) {
            float v   = bestv[rt][j];
            int   idx = besti[rt][j];
            #pragma unroll
            for (int m = 8; m >= 1; m >>= 1) {
                float ov = __shfl_xor(v, m, 32);
                int   oi = __shfl_xor(idx, m, 32);
                if (ov > v || (ov == v && oi < idx)) { v = ov; idx = oi; }
            }
            if (l16 == 0) {
                const int r = rt * 16 + j + (laneHi ? 8 : 0);
                red_v[r][wave] = v;
                red_i[r][wave] = idx;
            }
        }
    }
    __syncthreads();

    // Combine the 8 wave candidates per row; lowest index wins ties.
    if (tid < MBLK) {
        float bv = red_v[tid][0];
        int   bi = red_i[tid][0];
        #pragma unroll
        for (int w = 1; w < 8; ++w) {
            float v = red_v[tid][w];
            int   i = red_i[tid][w];
            if (v > bv || (v == bv && i < bi)) { bv = v; bi = i; }
        }
        out[row0 + tid] = bi;
    }
}

// ---------------------------------------------------------------------------
extern "C" void kernel_launch(void* const* d_in, const int* in_sizes, int n_in,
                              void* d_out, int out_size, void* d_ws, size_t ws_size,
                              hipStream_t stream) {
    (void)in_sizes; (void)n_in; (void)out_size; (void)ws_size;
    const float* embed   = (const float*)d_in[0];  // [16][1500][768] fp32
    const float* centers = (const float*)d_in[1];  // [1024][768] fp32
    int* out = (int*)d_out;                        // [16][1500] int32 indices

    // Workspace layout (~3.1 MB): B_hi | B_lo | c_half
    char*   ws     = (char*)d_ws;
    bf16_t* b_hi   = (bf16_t*)ws;
    bf16_t* b_lo   = (bf16_t*)(ws + (size_t)CN * DK * sizeof(bf16_t));
    float*  c_half = (float*)(ws + (size_t)CN * DK * 2 * sizeof(bf16_t));

    prep_centers_kernel<<<CN, 256, 0, stream>>>(centers, b_hi, b_lo, c_half);
    kmeans_argmax_kernel<<<M_TOT / MBLK, 256, 0, stream>>>(embed, b_hi, b_lo,
                                                           c_half, out);
}
